// LLaMAAttention_40982577938555
// MI455X (gfx1250) — compile-verified, matrix operand order checked by reading, not run on hardware
//
#include <hip/hip_runtime.h>
#include <hip/hip_bf16.h>

#define BB 4
#define SQ 2048
#define DMODEL 2048
#define HEADS 16
#define DH 128

typedef __attribute__((ext_vector_type(16))) __bf16 v16bf;
typedef __attribute__((ext_vector_type(8)))  float  v8f;
typedef __attribute__((ext_vector_type(4)))  unsigned int v4u;
typedef __attribute__((ext_vector_type(4)))  int v4i;
typedef __attribute__((ext_vector_type(8)))  int v8i;

// ---------- helpers ----------

__device__ __forceinline__ unsigned short f32_to_bf16(float f) {
    unsigned int u = __float_as_uint(f);
    u += 0x7FFFu + ((u >> 16) & 1u);          // round-to-nearest-even
    return (unsigned short)(u >> 16);
}
__device__ __forceinline__ float bf16_to_f32(unsigned short h) {
    return __uint_as_float(((unsigned int)h) << 16);
}
__device__ __forceinline__ v8f vzero8() {
    v8f v;
#pragma unroll
    for (int i = 0; i < 8; ++i) v[i] = 0.0f;
    return v;
}

// WMMA: D(f32 16x16) = A(bf16 16x32) * B(bf16 32x16) + C
__device__ __forceinline__ v8f wmma_bf16(v16bf a, v16bf b, v8f c) {
    return __builtin_amdgcn_wmma_f32_16x16x32_bf16(
        /*neg_a=*/false, a, /*neg_b=*/false, b,
        /*c_mod=*/(short)0, c, /*reuse_a=*/false, /*reuse_b=*/false);
}

// Load a 16x32 bf16 A-fragment (or B-fragment from an [N][K]-major tile).
// ISA 7.12.2: lane m (0-15): K=0..7 then 16..23 ; lane m+16: K=8..15 then 24..31.
__device__ __forceinline__ v16bf load_frag(const unsigned short* base, int lane, int stride) {
    const int m  = lane & 15;
    const int hi = (lane >> 4) & 1;
    const __bf16* p = reinterpret_cast<const __bf16*>(base + (size_t)m * stride + hi * 8);
    v16bf r;
#pragma unroll
    for (int i = 0; i < 8; ++i) {
        r[i]     = p[i];        // K = hi*8 + i
        r[i + 8] = p[i + 16];   // K = 16 + hi*8 + i
    }
    return r;
}

// ---------- Tensor Data Mover: 2D tile (16-bit elems) global -> LDS ----------
// D# per ISA ch.8: group0 = {count=1 | lds_addr | global_addr[56:0] | type=2},
// group1 = {data_size=2B, tensor_dim0/1, tile_dim0/1, tensor_dim0_stride}.
// Issue from ONE wave only (EXEC ignored, one instr = one DMA). TDM ops from a
// wave complete in order, so a partial s_wait_tensorcnt drains only the oldest.
__device__ __forceinline__ void tdm_load_2d(unsigned lds_byte_addr,
                                            const void* gsrc,
                                            unsigned tile_cols,            // dim0, elems
                                            unsigned tile_rows,            // dim1
                                            unsigned long long row_stride, // elems
                                            unsigned tensor_dim0,
                                            unsigned tensor_dim1) {
    unsigned long long ga = (unsigned long long)(size_t)gsrc;
    v4u g0;
    g0[0] = 1u;                                              // count=1, user mode
    g0[1] = lds_byte_addr;                                   // lds_addr (bytes)
    g0[2] = (unsigned)(ga & 0xFFFFFFFFull);                  // global_addr[31:0]
    g0[3] = ((unsigned)(ga >> 32) & 0x01FFFFFFu) | (2u << 30); // addr[56:32], type=2
    v8i g1;
    g1[0] = (int)(1u << 16);                                 // data_size = 1 -> 2 bytes
    g1[1] = (int)((tensor_dim0 & 0xFFFFu) << 16);            // td0[15:0] @ bits 63:48
    g1[2] = (int)((tensor_dim0 >> 16) | ((tensor_dim1 & 0xFFFFu) << 16));
    g1[3] = (int)((tensor_dim1 >> 16) | (tile_cols << 16));  // tile_dim0 @ 127:112
    g1[4] = (int)(tile_rows & 0xFFFFu);                      // tile_dim1; tile_dim2=0
    g1[5] = (int)(unsigned)(row_stride & 0xFFFFFFFFull);     // dim0_stride[31:0]
    g1[6] = (int)(unsigned)((row_stride >> 32) & 0xFFFFull); // dim0_stride[47:32]
    g1[7] = 0;
    v4i gz = {0, 0, 0, 0};
#if defined(__clang_major__) && (__clang_major__ >= 23)
    v8i gz8 = {0, 0, 0, 0, 0, 0, 0, 0};
    __builtin_amdgcn_tensor_load_to_lds(g0, g1, gz, gz, gz8, 0);
#else
    __builtin_amdgcn_tensor_load_to_lds(g0, g1, gz, gz, 0);
#endif
}

// ---------- elementwise kernels ----------

__global__ void cast_f32_to_bf16(const float* __restrict__ in,
                                 unsigned short* __restrict__ out, size_t n) {
    size_t i = (size_t)blockIdx.x * blockDim.x + threadIdx.x;
    size_t stride = (size_t)gridDim.x * blockDim.x;
    for (; i < n; i += stride) out[i] = f32_to_bf16(in[i]);
}

__global__ void rope_tables(float* __restrict__ cosT, float* __restrict__ sinT) {
    int idx = blockIdx.x * blockDim.x + threadIdx.x;   // S*DH threads
    int s = idx >> 7, d = idx & 127;
    float inv = powf(10000.0f, -((float)(2 * (d & 63))) * (1.0f / 128.0f));
    float f = (float)s * inv;
    cosT[idx] = cosf(f);
    sinT[idx] = sinf(f);
}

// In-place RoPE on q and k (bf16, [B,S,H*Dh]); one thread per (b,s,h,i<64) pair
__global__ void rope_apply(unsigned short* __restrict__ q,
                           unsigned short* __restrict__ k,
                           const float* __restrict__ cosT,
                           const float* __restrict__ sinT) {
    size_t idx = (size_t)blockIdx.x * blockDim.x + threadIdx.x;
    int i = (int)(idx & 63);
    int h = (int)((idx >> 6) & 15);
    int s = (int)((idx >> 10) & 2047);
    int b = (int)(idx >> 21);
    size_t base = ((size_t)(b * SQ + s)) * DMODEL + h * DH;
    float c  = cosT[s * DH + i];
    float sn = sinT[s * DH + i];
    {
        float x0 = bf16_to_f32(q[base + i]), x1 = bf16_to_f32(q[base + 64 + i]);
        q[base + i]      = f32_to_bf16(x0 * c - x1 * sn);
        q[base + 64 + i] = f32_to_bf16(x1 * c + x0 * sn);
    }
    {
        float x0 = bf16_to_f32(k[base + i]), x1 = bf16_to_f32(k[base + 64 + i]);
        k[base + i]      = f32_to_bf16(x0 * c - x1 * sn);
        k[base + 64 + i] = f32_to_bf16(x1 * c + x0 * sn);
    }
}

// ---------- WMMA GEMM:  C[M,N] = A[M,K] @ W[N,K]^T  (nn.Linear) ----------
// Block: 128 thr = 4 waves, tile 128(M) x 64(N); wave owns a 32x64 strip.
// K chunks of 32, double-buffered TDM staging overlapped with WMMA compute.
template <bool OUT_BF16>
__global__ __launch_bounds__(128) void gemm_wmma(const unsigned short* __restrict__ A,
                                                 const unsigned short* __restrict__ W,
                                                 void* __restrict__ out,
                                                 int M, int N, int K) {
    __shared__ unsigned short As[2][128 * 32];
    __shared__ unsigned short Bs[2][64 * 32];

    const int tid  = threadIdx.x;
    const int lane = tid & 31;
    const int wid  = tid >> 5;
    const int hi   = (lane >> 4) & 1;
    const int n16  = lane & 15;
    const int m0   = blockIdx.y * 128;
    const int n0   = blockIdx.x * 64;

    v8f acc0[4], acc1[4];
#pragma unroll
    for (int t = 0; t < 4; ++t) { acc0[t] = vzero8(); acc1[t] = vzero8(); }

    const unsigned ldsA[2] = { (unsigned)(size_t)(void*)As[0], (unsigned)(size_t)(void*)As[1] };
    const unsigned ldsB[2] = { (unsigned)(size_t)(void*)Bs[0], (unsigned)(size_t)(void*)Bs[1] };

    // prologue: kick off chunk 0 into buffer 0
    if (wid == 0) {
        tdm_load_2d(ldsA[0], A + (size_t)m0 * K, 32, 128,
                    (unsigned long long)K, (unsigned)K, (unsigned)M);
        tdm_load_2d(ldsB[0], W + (size_t)n0 * K, 32, 64,
                    (unsigned long long)K, (unsigned)K, (unsigned)N);
    }

    int cur = 0;
    for (int kc = 0; kc < K; kc += 32, cur ^= 1) {
        if (wid == 0) {
            if (kc + 32 < K) {
                // stream next chunk into the other buffer, then drain only the
                // current pair (TDM completes in order): compute overlaps DMA.
                tdm_load_2d(ldsA[cur ^ 1], A + (size_t)m0 * K + kc + 32, 32, 128,
                            (unsigned long long)K, (unsigned)K, (unsigned)M);
                tdm_load_2d(ldsB[cur ^ 1], W + (size_t)n0 * K + kc + 32, 32, 64,
                            (unsigned long long)K, (unsigned)K, (unsigned)N);
                __builtin_amdgcn_s_wait_tensorcnt(2);
            } else {
                __builtin_amdgcn_s_wait_tensorcnt(0);
            }
        } else if (kc + 32 < K) {
            // warm GL2 for the next chunk while wave 0 drives the TDM
            __builtin_prefetch(A + (size_t)(m0 + tid) * K + kc + 32, 0, 1);
        }
        __syncthreads();

        const unsigned short* Ac = As[cur];
        const unsigned short* Bc = Bs[cur];
        v16bf af0 = load_frag(&Ac[(wid * 32) * 32], lane, 32);
        v16bf af1 = load_frag(&Ac[(wid * 32 + 16) * 32], lane, 32);
#pragma unroll
        for (int t = 0; t < 4; ++t) {
            v16bf bf = load_frag(&Bc[(t * 16) * 32], lane, 32);
            acc0[t] = wmma_bf16(af0, bf, acc0[t]);
            acc1[t] = wmma_bf16(af1, bf, acc1[t]);
        }
        __syncthreads();
    }

    // epilogue: C layout — lane n = lane&15, VGPR r -> row r + 8*hi
#pragma unroll
    for (int t = 0; t < 4; ++t) {
#pragma unroll
        for (int r = 0; r < 8; ++r) {
            size_t row0 = (size_t)(m0 + wid * 32 + r + 8 * hi);
            size_t col  = (size_t)(n0 + t * 16 + n16);
            if (OUT_BF16) {
                ((unsigned short*)out)[row0 * N + col]        = f32_to_bf16(acc0[t][r]);
                ((unsigned short*)out)[(row0 + 16) * N + col] = f32_to_bf16(acc1[t][r]);
            } else {
                ((float*)out)[row0 * N + col]        = acc0[t][r];
                ((float*)out)[(row0 + 16) * N + col] = acc1[t][r];
            }
        }
    }
}

// ---------- Flash attention (causal), bf16 in/out, f32 online softmax ----------
// grid = (S/64, B*H), block = 128 (4 waves); wave w owns q rows q0+16w..+15.
// K tiles double-buffered via TDM; V staged transposed by the block.
__global__ __launch_bounds__(128) void attn_kernel(const unsigned short* __restrict__ qb,
                                                   const unsigned short* __restrict__ kb,
                                                   const unsigned short* __restrict__ vb,
                                                   unsigned short* __restrict__ ob) {
    __shared__ unsigned short Ks[2][32 * 128];  // K chunk, [k_local][d] (TDM-staged)
    __shared__ unsigned short Vts[128 * 32];    // V chunk transposed, [d][k_local]
    __shared__ unsigned short Ps[4][16 * 32];   // per-wave P tile [m][k_local]

    const int tid  = threadIdx.x;
    const int lane = tid & 31;
    const int wid  = tid >> 5;
    const int hi   = (lane >> 4) & 1;
    const int n16  = lane & 15;

    const int q0 = blockIdx.x * 64;
    const int bh = blockIdx.y;
    const int b  = bh >> 4;
    const int h  = bh & 15;
    const int qw = q0 + wid * 16;
    const size_t rowbase = (size_t)b * SQ;
    const float SCALE = 0.08838834764831845f;   // 1/sqrt(128)
    const int kc_max = q0 + 32;                 // last chunk start (causal)

    const unsigned ldsK[2] = { (unsigned)(size_t)(void*)Ks[0], (unsigned)(size_t)(void*)Ks[1] };

    // prologue: kick off K chunk 0
    if (wid == 0) {
        tdm_load_2d(ldsK[0], kb + rowbase * DMODEL + h * DH,
                    128, 32, (unsigned long long)DMODEL,
                    (unsigned)DMODEL, (unsigned)(BB * SQ));
    }

    // Q fragments pinned in registers for the whole K loop
    v16bf qfrag[4];
    {
        const unsigned short* qptr = qb + (rowbase + qw) * (size_t)DMODEL + h * DH;
#pragma unroll
        for (int dc = 0; dc < 4; ++dc)
            qfrag[dc] = load_frag(qptr + dc * 32, lane, DMODEL);
    }

    v8f Of[8];
#pragma unroll
    for (int t = 0; t < 8; ++t) Of[t] = vzero8();
    float mprev[8], lrow[8];
#pragma unroll
    for (int r = 0; r < 8; ++r) { mprev[r] = -3.0e38f; lrow[r] = 0.0f; }

    int cur = 0;
    for (int kc = 0; kc <= kc_max; kc += 32, cur ^= 1) {
        // ---- stage V chunk transposed -> Vts[d][k] (all threads)
        {
            int kl = tid & 31, dg = tid >> 5;
            const unsigned short* src = vb + (rowbase + kc + kl) * (size_t)DMODEL + h * DH + dg * 32;
#pragma unroll
            for (int i = 0; i < 32; ++i) Vts[(dg * 32 + i) * 32 + kl] = src[i];
        }
        // ---- K[cur^1] <- next chunk via TDM, then drain only K[cur]
        if (wid == 0) {
            if (kc + 32 <= kc_max) {
                tdm_load_2d(ldsK[cur ^ 1], kb + (rowbase + kc + 32) * (size_t)DMODEL + h * DH,
                            128, 32, (unsigned long long)DMODEL,
                            (unsigned)DMODEL, (unsigned)(BB * SQ));
                __builtin_amdgcn_s_wait_tensorcnt(1);
            } else {
                __builtin_amdgcn_s_wait_tensorcnt(0);
            }
        }
        __syncthreads();

        if (kc <= qw + 15) {                 // wave-uniform: EXEC stays all-ones
            const bool needmask = (kc + 31 > qw);
            const unsigned short* Kc = Ks[cur];

            // scores: two 16-wide k sub-tiles, K-dim = 128 (4 chained wmmas)
            v8f s0 = vzero8(), s1 = vzero8();
#pragma unroll
            for (int dc = 0; dc < 4; ++dc) {
                v16bf kf0 = load_frag(&Kc[dc * 32], lane, 128);
                v16bf kf1 = load_frag(&Kc[16 * 128 + dc * 32], lane, 128);
                s0 = wmma_bf16(qfrag[dc], kf0, s0);
                s1 = wmma_bf16(qfrag[dc], kf1, s1);
            }

            float p0[8], p1[8], mcur[8];
#pragma unroll
            for (int r = 0; r < 8; ++r) {
                const int q_idx = qw + r + 8 * hi;
                float a0 = s0[r] * SCALE;
                float a1 = s1[r] * SCALE;
                if (needmask) {
                    if (kc + n16 > q_idx)      a0 = -3.0e38f;
                    if (kc + 16 + n16 > q_idx) a1 = -3.0e38f;
                }
                p0[r] = a0; p1[r] = a1;
                mcur[r] = fmaxf(a0, a1);
            }
#pragma unroll
            for (int off = 1; off < 16; off <<= 1)
#pragma unroll
                for (int r = 0; r < 8; ++r)
                    mcur[r] = fmaxf(mcur[r], __shfl_xor(mcur[r], off, 32));

            float alpha[8], rsum[8];
#pragma unroll
            for (int r = 0; r < 8; ++r) {
                float mn = fmaxf(mprev[r], mcur[r]);
                alpha[r] = __expf(mprev[r] - mn);
                mprev[r] = mn;
                p0[r] = __expf(p0[r] - mn);
                p1[r] = __expf(p1[r] - mn);
                rsum[r] = p0[r] + p1[r];
            }
#pragma unroll
            for (int off = 1; off < 16; off <<= 1)
#pragma unroll
                for (int r = 0; r < 8; ++r)
                    rsum[r] += __shfl_xor(rsum[r], off, 32);
#pragma unroll
            for (int r = 0; r < 8; ++r)
                lrow[r] = lrow[r] * alpha[r] + rsum[r];

            // rescale O accumulator
#pragma unroll
            for (int t = 0; t < 8; ++t)
#pragma unroll
                for (int r = 0; r < 8; ++r)
                    Of[t][r] *= alpha[r];

            // C-layout P -> LDS -> A-fragment
            unsigned short* Pw = Ps[wid];
#pragma unroll
            for (int r = 0; r < 8; ++r) {
                int m = r + 8 * hi;
                Pw[m * 32 + n16]      = f32_to_bf16(p0[r]);
                Pw[m * 32 + 16 + n16] = f32_to_bf16(p1[r]);
            }
            asm volatile("s_wait_dscnt 0" ::: "memory");   // wave-local LDS RAW
            v16bf pf = load_frag(Pw, lane, 32);
#pragma unroll
            for (int t = 0; t < 8; ++t) {
                v16bf vf = load_frag(&Vts[(t * 16) * 32], lane, 32);
                Of[t] = wmma_bf16(pf, vf, Of[t]);
            }
        }
        __syncthreads();
    }

    // normalize + store bf16 into [B,S,H*Dh]
    float invl[8];
#pragma unroll
    for (int r = 0; r < 8; ++r) invl[r] = 1.0f / lrow[r];
#pragma unroll
    for (int t = 0; t < 8; ++t) {
#pragma unroll
        for (int r = 0; r < 8; ++r) {
            int qi = qw + r + 8 * hi;
            size_t o = (rowbase + qi) * (size_t)DMODEL + h * DH + t * 16 + n16;
            ob[o] = f32_to_bf16(Of[t][r] * invl[r]);
        }
    }
}

// ---------- host ----------

extern "C" void kernel_launch(void* const* d_in, const int* in_sizes, int n_in,
                              void* d_out, int out_size, void* d_ws, size_t ws_size,
                              hipStream_t stream) {
    const float* x  = (const float*)d_in[0];
    // d_in[1] = attn_mask (causal, known statically) — unused
    const float* Wq = (const float*)d_in[2];
    const float* Wk = (const float*)d_in[3];
    const float* Wv = (const float*)d_in[4];
    const float* Wo = (const float*)d_in[5];
    float* out = (float*)d_out;

    char* ws = (char*)d_ws;
    size_t off = 0;
    auto carve = [&](size_t bytes) -> void* {
        void* p = ws + off;
        off += (bytes + 255) & ~(size_t)255;
        return p;
    };
    const size_t NX = (size_t)BB * SQ * DMODEL;   // 16,777,216
    const size_t NW = (size_t)DMODEL * DMODEL;    //  4,194,304

    float*          cosT = (float*)carve((size_t)SQ * DH * 4);
    float*          sinT = (float*)carve((size_t)SQ * DH * 4);
    unsigned short* xb   = (unsigned short*)carve(NX * 2);
    unsigned short* wqb  = (unsigned short*)carve(NW * 2);
    unsigned short* wkb  = (unsigned short*)carve(NW * 2);
    unsigned short* wvb  = (unsigned short*)carve(NW * 2);
    unsigned short* wob  = (unsigned short*)carve(NW * 2);
    unsigned short* qbuf = (unsigned short*)carve(NX * 2);
    unsigned short* kbuf = (unsigned short*)carve(NX * 2);
    unsigned short* vbuf = (unsigned short*)carve(NX * 2);
    unsigned short* abuf = (unsigned short*)carve(NX * 2);

    rope_tables<<<(SQ * DH) / 256, 256, 0, stream>>>(cosT, sinT);
    cast_f32_to_bf16<<<2048, 256, 0, stream>>>(x, xb, NX);
    cast_f32_to_bf16<<<1024, 256, 0, stream>>>(Wq, wqb, NW);
    cast_f32_to_bf16<<<1024, 256, 0, stream>>>(Wk, wkb, NW);
    cast_f32_to_bf16<<<1024, 256, 0, stream>>>(Wv, wvb, NW);
    cast_f32_to_bf16<<<1024, 256, 0, stream>>>(Wo, wob, NW);

    dim3 gp(DMODEL / 64, (BB * SQ) / 128);   // (32, 64)
    gemm_wmma<true><<<gp, 128, 0, stream>>>(xb, wqb, (void*)qbuf, BB * SQ, DMODEL, DMODEL);
    gemm_wmma<true><<<gp, 128, 0, stream>>>(xb, wkb, (void*)kbuf, BB * SQ, DMODEL, DMODEL);
    gemm_wmma<true><<<gp, 128, 0, stream>>>(xb, wvb, (void*)vbuf, BB * SQ, DMODEL, DMODEL);

    rope_apply<<<((size_t)BB * SQ * HEADS * 64) / 256, 256, 0, stream>>>(qbuf, kbuf, cosT, sinT);

    attn_kernel<<<dim3(SQ / 64, BB * HEADS), 128, 0, stream>>>(qbuf, kbuf, vbuf, abuf);

    gemm_wmma<false><<<gp, 128, 0, stream>>>(abuf, wob, (void*)out, BB * SQ, DMODEL, DMODEL);
}
